// MoE_38843684225093
// MI455X (gfx1250) — matrix operand order repaired and checked by reading, not compiled, not run on hardware
//
#include <hip/hip_runtime.h>
#include <stdint.h>

// ---------------------------------------------------------------------------
// MoE top-2 router + grouped GEMM for MI455X (gfx1250, wave32, WMMA).
// Math path: bf16x3 split precision on v_wmma_f32_16x16x32_bf16 (fp32 acc).
// ---------------------------------------------------------------------------

#define D      1024
#define E      8
#define BM     128      // M tile (tokens) per workgroup
#define BN     128      // N tile (hidden) per workgroup
#define BK     32       // K per LDS stage == one WMMA K-step
#define LDSTR  40       // padded LDS row stride (elements) to spread banks

typedef __attribute__((ext_vector_type(16))) __bf16        v16bf;
typedef __attribute__((ext_vector_type(8)))  float         v8f;
typedef __attribute__((ext_vector_type(4)))  unsigned int  u32x4;

union OpBf { u32x4 q[2]; v16bf v; };

__device__ __forceinline__ unsigned short f2bf_rne(float f) {
  unsigned u = __float_as_uint(f);
  unsigned r = u + 0x7FFFu + ((u >> 16) & 1u);
  return (unsigned short)(r >> 16);
}
__device__ __forceinline__ float bf2f(unsigned short h) {
  return __uint_as_float(((unsigned)h) << 16);
}

// ---------------------------------------------------------------------------
// zero small int arrays
__global__ void k_zero32(int* __restrict__ p, int n) {
  int i = blockIdx.x * blockDim.x + threadIdx.x;
  if (i < n) p[i] = 0;
}

// ---------------------------------------------------------------------------
// Gating: one wave (32 lanes) per token. logits = x.gW^T + gb, softmax,
// top-2 (lowest index wins ties, matching jax), renormalize via softmax of
// the two probabilities (matches reference exactly).
__global__ void k_gate(const float* __restrict__ x, const float* __restrict__ gW,
                       const float* __restrict__ gb, float* __restrict__ gprob,
                       int* __restrict__ topi, float* __restrict__ topw,
                       int* __restrict__ counts, int N) {
  int wid  = threadIdx.x >> 5;
  int lane = threadIdx.x & 31;
  int tok  = blockIdx.x * 8 + wid;
  if (tok >= N) return;
  const float* xr = x + (size_t)tok * D;

  float acc[E];
#pragma unroll
  for (int e = 0; e < E; ++e) acc[e] = 0.0f;

  for (int d = lane; d < D; d += 32) {
    float xv = xr[d];
#pragma unroll
    for (int e = 0; e < E; ++e) acc[e] += xv * gW[e * D + d];
  }
#pragma unroll
  for (int e = 0; e < E; ++e) {
    float v = acc[e];
#pragma unroll
    for (int o = 16; o > 0; o >>= 1) v += __shfl_xor(v, o, 32);
    acc[e] = v + gb[e];
  }
  // softmax over 8 (redundant in all lanes; registers only)
  float m = acc[0];
#pragma unroll
  for (int e = 1; e < E; ++e) m = fmaxf(m, acc[e]);
  float s = 0.0f;
#pragma unroll
  for (int e = 0; e < E; ++e) { acc[e] = __expf(acc[e] - m); s += acc[e]; }
  float inv = 1.0f / s;
#pragma unroll
  for (int e = 0; e < E; ++e) acc[e] *= inv;

  // top-2 scan (strict > keeps lowest index on ties)
  float v0 = -1.0f, v1 = -1.0f;
  int   i0 = 0,     i1 = 0;
#pragma unroll
  for (int e = 0; e < E; ++e) {
    float pe = acc[e];
    if (pe > v0)      { v1 = v0; i1 = i0; v0 = pe; i0 = e; }
    else if (pe > v1) { v1 = pe; i1 = e; }
  }
  if (lane == 0) {
#pragma unroll
    for (int e = 0; e < E; ++e) gprob[(size_t)tok * E + e] = acc[e];
    float e1 = __expf(v1 - v0);
    float w0 = 1.0f / (1.0f + e1);
    topi[tok * 2]     = i0;  topi[tok * 2 + 1] = i1;
    topw[tok * 2]     = w0;  topw[tok * 2 + 1] = 1.0f - w0;
    atomicAdd(&counts[i0], 1);
    atomicAdd(&counts[i1], 1);
  }
}

// ---------------------------------------------------------------------------
// Expert bases aligned to BM so GEMM tiles never cross experts; init all
// bucket slots to token 0 so padded rows are safe garbage.
__global__ void k_offsets(const int* __restrict__ counts, int* __restrict__ base,
                          int* __restrict__ btok, int rows) {
  if (threadIdx.x == 0) {
    int a = 0;
    for (int e = 0; e < E; ++e) { base[e] = a; a += ((counts[e] + BM - 1) / BM) * BM; }
  }
  for (int i = threadIdx.x; i < rows; i += blockDim.x) btok[i] = 0;
}

__global__ void k_fill(const int* __restrict__ topi, const int* __restrict__ base,
                       int* __restrict__ counts2, int* __restrict__ btok,
                       int* __restrict__ slot_of, int N) {
  int t = blockIdx.x * blockDim.x + threadIdx.x;
  if (t >= N) return;
#pragma unroll
  for (int k = 0; k < 2; ++k) {
    int e   = topi[t * 2 + k];
    int pos = atomicAdd(&counts2[e], 1);
    int s   = base[e] + pos;
    btok[s] = t;
    slot_of[t * 2 + k] = s;
  }
}

// ---------------------------------------------------------------------------
// fp32 -> (bf16 hi, bf16 lo) planes, one float4 per thread.
__global__ void k_split(const float* __restrict__ src, unsigned short* __restrict__ hi,
                        unsigned short* __restrict__ lo, int n4) {
  int i = blockIdx.x * blockDim.x + threadIdx.x;
  if (i >= n4) return;
  float4 v = ((const float4*)src)[i];
  float f[4] = {v.x, v.y, v.z, v.w};
  unsigned short h[4], l[4];
#pragma unroll
  for (int j = 0; j < 4; ++j) {
    h[j] = f2bf_rne(f[j]);
    l[j] = f2bf_rne(f[j] - bf2f(h[j]));
  }
  uint2 hv; hv.x = (unsigned)h[0] | ((unsigned)h[1] << 16); hv.y = (unsigned)h[2] | ((unsigned)h[3] << 16);
  uint2 lv; lv.x = (unsigned)l[0] | ((unsigned)l[1] << 16); lv.y = (unsigned)l[2] | ((unsigned)l[3] << 16);
  ((uint2*)hi)[i] = hv;
  ((uint2*)lo)[i] = lv;
}

// ---------------------------------------------------------------------------
// Shared compute core: 8 waves as 4(M) x 2(N); each wave owns 32x64 = 2x4
// tiles of 16x16 fp32 accumulators. Per K-step: 24 v_wmma (hh + hl + lh).
#define WMMA_BF16(A, B, C) \
  __builtin_amdgcn_wmma_f32_16x16x32_bf16(false, (A), false, (B), (short)0, (C), false, false)

__device__ __forceinline__ void gemm_core(const unsigned short* sAhi, const unsigned short* sAlo,
                                          const unsigned short* sBhi, const unsigned short* sBlo,
                                          int wave_m, int wave_n, int lr, int hh,
                                          v8f acc[2][4]) {
  OpBf aH[2], aL[2], bH[4], bL[4];
#pragma unroll
  for (int mt = 0; mt < 2; ++mt) {
    int r = (wave_m * 32 + mt * 16 + lr) * LDSTR + hh * 16;  // A: one contiguous 32B chunk
    aH[mt].q[0] = *(const u32x4*)&sAhi[r];
    aH[mt].q[1] = *(const u32x4*)&sAhi[r + 8];
    aL[mt].q[0] = *(const u32x4*)&sAlo[r];
    aL[mt].q[1] = *(const u32x4*)&sAlo[r + 8];
  }
#pragma unroll
  for (int nt = 0; nt < 4; ++nt) {
    int r = (wave_n * 64 + nt * 16 + lr) * LDSTR + hh * 16;  // B: one contiguous 32B chunk
    bH[nt].q[0] = *(const u32x4*)&sBhi[r];
    bH[nt].q[1] = *(const u32x4*)&sBhi[r + 8];
    bL[nt].q[0] = *(const u32x4*)&sBlo[r];
    bL[nt].q[1] = *(const u32x4*)&sBlo[r + 8];
  }
#pragma unroll
  for (int mt = 0; mt < 2; ++mt)
#pragma unroll
    for (int nt = 0; nt < 4; ++nt) {
      v8f c = acc[mt][nt];
      c = WMMA_BF16(aH[mt].v, bH[nt].v, c);
      c = WMMA_BF16(aH[mt].v, bL[nt].v, c);
      c = WMMA_BF16(aL[mt].v, bH[nt].v, c);
      acc[mt][nt] = c;
    }
}

// ---------------------------------------------------------------------------
// GEMM1: H = relu(Xg . W1^T + b1)  (gather fp32 x, split to hi/lo in LDS)
__global__ __launch_bounds__(256) void k_gemm1(
    const float* __restrict__ x,
    const unsigned short* __restrict__ w1hi, const unsigned short* __restrict__ w1lo,
    const float* __restrict__ b1,
    const int* __restrict__ counts, const int* __restrict__ base,
    const int* __restrict__ btok,
    unsigned short* __restrict__ Hhi, unsigned short* __restrict__ Hlo) {
  int e = blockIdx.x, mtile = blockIdx.y;
  if (mtile * BM >= counts[e]) return;          // uniform early exit
  int n0 = blockIdx.z * BN;
  int rowBase = base[e] + mtile * BM;

  __shared__ unsigned short sAhi[BM * LDSTR], sAlo[BM * LDSTR];
  __shared__ unsigned short sBhi[BN * LDSTR], sBlo[BN * LDSTR];

  int tid = threadIdx.x, lane = tid & 31, wid = tid >> 5;
  int wave_m = wid >> 1, wave_n = wid & 1;
  int lr = lane & 15, hh = lane >> 4;

  int srow = tid >> 1;                // 0..127
  int skh  = (tid & 1) * 16;          // 0 / 16
  int tokr = btok[rowBase + srow];
  const float* xrow = x + (size_t)tokr * D;
  const unsigned short* wHiRow = w1hi + ((size_t)e * D + (size_t)(n0 + srow)) * D;
  const unsigned short* wLoRow = w1lo + ((size_t)e * D + (size_t)(n0 + srow)) * D;

  v8f acc[2][4];
  const v8f zacc = {0.f, 0.f, 0.f, 0.f, 0.f, 0.f, 0.f, 0.f};
#pragma unroll
  for (int mt = 0; mt < 2; ++mt)
#pragma unroll
    for (int nt = 0; nt < 4; ++nt) acc[mt][nt] = zacc;

  for (int k0 = 0; k0 < D; k0 += BK) {
    // stage A: gather 16 fp32, split to hi/lo
#pragma unroll
    for (int j = 0; j < 4; ++j) {
      float4 v = *(const float4*)(xrow + k0 + skh + j * 4);
      float f[4] = {v.x, v.y, v.z, v.w};
      unsigned short h[4], l[4];
#pragma unroll
      for (int q = 0; q < 4; ++q) { h[q] = f2bf_rne(f[q]); l[q] = f2bf_rne(f[q] - bf2f(h[q])); }
      uint2 hv; hv.x = (unsigned)h[0] | ((unsigned)h[1] << 16); hv.y = (unsigned)h[2] | ((unsigned)h[3] << 16);
      uint2 lv; lv.x = (unsigned)l[0] | ((unsigned)l[1] << 16); lv.y = (unsigned)l[2] | ((unsigned)l[3] << 16);
      *(uint2*)&sAhi[srow * LDSTR + skh + j * 4] = hv;
      *(uint2*)&sAlo[srow * LDSTR + skh + j * 4] = lv;
    }
    // stage B: pre-split planes, straight copies
    *(u32x4*)&sBhi[srow * LDSTR + skh]     = *(const u32x4*)(wHiRow + k0 + skh);
    *(u32x4*)&sBhi[srow * LDSTR + skh + 8] = *(const u32x4*)(wHiRow + k0 + skh + 8);
    *(u32x4*)&sBlo[srow * LDSTR + skh]     = *(const u32x4*)(wLoRow + k0 + skh);
    *(u32x4*)&sBlo[srow * LDSTR + skh + 8] = *(const u32x4*)(wLoRow + k0 + skh + 8);
    __syncthreads();
    gemm_core(sAhi, sAlo, sBhi, sBlo, wave_m, wave_n, lr, hh, acc);
    __syncthreads();
  }

  // bias + relu + split-store H (hi/lo planes, row-major [slot][k])
#pragma unroll
  for (int nt = 0; nt < 4; ++nt) {
    int ncol = n0 + wave_n * 64 + nt * 16 + lr;
    float bias = b1[e * D + ncol];
#pragma unroll
    for (int mt = 0; mt < 2; ++mt) {
      int mrow0 = rowBase + wave_m * 32 + mt * 16 + hh * 8;
#pragma unroll
      for (int i = 0; i < 8; ++i) {
        float v = fmaxf(acc[mt][nt][i] + bias, 0.0f);
        unsigned short h = f2bf_rne(v);
        unsigned short l = f2bf_rne(v - bf2f(h));
        size_t o = (size_t)(mrow0 + i) * D + ncol;
        Hhi[o] = h;  Hlo[o] = l;
      }
    }
  }
}

// ---------------------------------------------------------------------------
// GEMM2: contrib = H . W2^T + b2   (A already split, contiguous slot rows)
__global__ __launch_bounds__(256) void k_gemm2(
    const unsigned short* __restrict__ Hhi, const unsigned short* __restrict__ Hlo,
    const unsigned short* __restrict__ w2hi, const unsigned short* __restrict__ w2lo,
    const float* __restrict__ b2,
    const int* __restrict__ counts, const int* __restrict__ base,
    float* __restrict__ contrib) {
  int e = blockIdx.x, mtile = blockIdx.y;
  if (mtile * BM >= counts[e]) return;
  int n0 = blockIdx.z * BN;
  int rowBase = base[e] + mtile * BM;

  __shared__ unsigned short sAhi[BM * LDSTR], sAlo[BM * LDSTR];
  __shared__ unsigned short sBhi[BN * LDSTR], sBlo[BN * LDSTR];

  int tid = threadIdx.x, lane = tid & 31, wid = tid >> 5;
  int wave_m = wid >> 1, wave_n = wid & 1;
  int lr = lane & 15, hh = lane >> 4;

  int srow = tid >> 1;
  int skh  = (tid & 1) * 16;
  const unsigned short* aHiRow = Hhi + (size_t)(rowBase + srow) * D;
  const unsigned short* aLoRow = Hlo + (size_t)(rowBase + srow) * D;
  const unsigned short* wHiRow = w2hi + ((size_t)e * D + (size_t)(n0 + srow)) * D;
  const unsigned short* wLoRow = w2lo + ((size_t)e * D + (size_t)(n0 + srow)) * D;

  v8f acc[2][4];
  const v8f zacc = {0.f, 0.f, 0.f, 0.f, 0.f, 0.f, 0.f, 0.f};
#pragma unroll
  for (int mt = 0; mt < 2; ++mt)
#pragma unroll
    for (int nt = 0; nt < 4; ++nt) acc[mt][nt] = zacc;

  for (int k0 = 0; k0 < D; k0 += BK) {
    *(u32x4*)&sAhi[srow * LDSTR + skh]     = *(const u32x4*)(aHiRow + k0 + skh);
    *(u32x4*)&sAhi[srow * LDSTR + skh + 8] = *(const u32x4*)(aHiRow + k0 + skh + 8);
    *(u32x4*)&sAlo[srow * LDSTR + skh]     = *(const u32x4*)(aLoRow + k0 + skh);
    *(u32x4*)&sAlo[srow * LDSTR + skh + 8] = *(const u32x4*)(aLoRow + k0 + skh + 8);
    *(u32x4*)&sBhi[srow * LDSTR + skh]     = *(const u32x4*)(wHiRow + k0 + skh);
    *(u32x4*)&sBhi[srow * LDSTR + skh + 8] = *(const u32x4*)(wHiRow + k0 + skh + 8);
    *(u32x4*)&sBlo[srow * LDSTR + skh]     = *(const u32x4*)(wLoRow + k0 + skh);
    *(u32x4*)&sBlo[srow * LDSTR + skh + 8] = *(const u32x4*)(wLoRow + k0 + skh + 8);
    __syncthreads();
    gemm_core(sAhi, sAlo, sBhi, sBlo, wave_m, wave_n, lr, hh, acc);
    __syncthreads();
  }

#pragma unroll
  for (int nt = 0; nt < 4; ++nt) {
    int ncol = n0 + wave_n * 64 + nt * 16 + lr;
    float bias = b2[e * D + ncol];
#pragma unroll
    for (int mt = 0; mt < 2; ++mt) {
      int mrow0 = rowBase + wave_m * 32 + mt * 16 + hh * 8;
#pragma unroll
      for (int i = 0; i < 8; ++i)
        contrib[(size_t)(mrow0 + i) * D + ncol] = acc[mt][nt][i] + bias;
    }
  }
}

// ---------------------------------------------------------------------------
// y[t] = w0 * contrib[slot0] + w1 * contrib[slot1]   (one block per token)
__global__ void k_combine(const float* __restrict__ contrib, const int* __restrict__ slot_of,
                          const float* __restrict__ topw, float* __restrict__ y) {
  int t  = blockIdx.x;
  int s0 = slot_of[t * 2], s1 = slot_of[t * 2 + 1];
  float w0 = topw[t * 2],  w1 = topw[t * 2 + 1];
  const float4* c0 = (const float4*)(contrib + (size_t)s0 * D);
  const float4* c1 = (const float4*)(contrib + (size_t)s1 * D);
  float4* yr = (float4*)(y + (size_t)t * D);
  int c = threadIdx.x;                 // 256 threads, D/4 = 256 float4
  float4 a = c0[c], b = c1[c], r;
  r.x = w0 * a.x + w1 * b.x;  r.y = w0 * a.y + w1 * b.y;
  r.z = w0 * a.z + w1 * b.z;  r.w = w0 * a.w + w1 * b.w;
  yr[c] = r;
}

// ---------------------------------------------------------------------------
extern "C" void kernel_launch(void* const* d_in, const int* in_sizes, int n_in,
                              void* d_out, int out_size, void* d_ws, size_t ws_size,
                              hipStream_t stream) {
  const float* x  = (const float*)d_in[0];
  const float* gW = (const float*)d_in[1];
  const float* gb = (const float*)d_in[2];
  const float* w1 = (const float*)d_in[3];
  const float* b1 = (const float*)d_in[4];
  const float* w2 = (const float*)d_in[5];
  const float* b2 = (const float*)d_in[6];

  const int N    = in_sizes[0] / D;         // 4096 tokens
  const int rows = 2 * N + E * BM;          // slot capacity (BM-aligned per expert)

  char* ws = (char*)d_ws;
  size_t off = 0;
  auto alloc = [&](size_t bytes) {
    size_t o = off;
    off += (bytes + 255) & ~(size_t)255;
    return o;
  };
  int*   counts  = (int*)(ws + alloc(E * 4));
  int*   counts2 = (int*)(ws + alloc(E * 4));
  int*   base    = (int*)(ws + alloc(E * 4));
  int*   topi    = (int*)(ws + alloc((size_t)N * 2 * 4));
  float* topw    = (float*)(ws + alloc((size_t)N * 2 * 4));
  int*   slot_of = (int*)(ws + alloc((size_t)N * 2 * 4));
  int*   btok    = (int*)(ws + alloc((size_t)rows * 4));
  unsigned short* w1hi = (unsigned short*)(ws + alloc((size_t)E * D * D * 2));
  unsigned short* w1lo = (unsigned short*)(ws + alloc((size_t)E * D * D * 2));
  unsigned short* w2hi = (unsigned short*)(ws + alloc((size_t)E * D * D * 2));
  unsigned short* w2lo = (unsigned short*)(ws + alloc((size_t)E * D * D * 2));
  unsigned short* Hhi  = (unsigned short*)(ws + alloc((size_t)rows * D * 2));
  unsigned short* Hlo  = (unsigned short*)(ws + alloc((size_t)rows * D * 2));
  float* contrib       = (float*)(ws + alloc((size_t)rows * D * 4));

  float* y     = (float*)d_out;
  float* gprob = y + (size_t)N * D;

  k_zero32<<<1, 32, 0, stream>>>(counts, E);
  k_zero32<<<1, 32, 0, stream>>>(counts2, E);
  k_gate<<<(N + 7) / 8, 256, 0, stream>>>(x, gW, gb, gprob, topi, topw, counts, N);
  k_offsets<<<1, 256, 0, stream>>>(counts, base, btok, rows);
  k_fill<<<(N + 255) / 256, 256, 0, stream>>>(topi, base, counts2, btok, slot_of, N);

  int n4 = (E * D * D) / 4;
  k_split<<<(n4 + 255) / 256, 256, 0, stream>>>(w1, w1hi, w1lo, n4);
  k_split<<<(n4 + 255) / 256, 256, 0, stream>>>(w2, w2hi, w2lo, n4);

  dim3 gg(E, (unsigned)((N + BM - 1) / BM), D / BN);
  k_gemm1<<<gg, 256, 0, stream>>>(x, w1hi, w1lo, b1, counts, base, btok, Hhi, Hlo);
  k_gemm2<<<gg, 256, 0, stream>>>(Hhi, Hlo, w2hi, w2lo, b2, counts, base, contrib);
  k_combine<<<N, 256, 0, stream>>>(contrib, slot_of, topw, y);
}
